// CausalFlowModel_32289564131532
// MI455X (gfx1250) — matrix operand order repaired and checked by reading, not compiled, not run on hardware
//
#include <hip/hip_runtime.h>
#include <hip/hip_bf16.h>
#include <cmath>
#include <cstdint>

// ---------------------------------------------------------------------------
// CDNA5 / gfx1250 WMMA pipeline for CausalFlowModel.
//  - All matmuls: v_wmma_f32_16x16x32_bf16 (wave32).
//  - RNN scan (511 dependent steps) in ONE persistent kernel:
//      * 16 blocks (128x128 tiles); per-ROW-GROUP (4-block) atomic barriers
//      * i2h weight slice resident in LDS (320KB/WGP) for all steps
//      * H_prev tile and bf16-converted u_t tile staged to LDS per step;
//        all WMMA operands via ds_load_b128 in steady state
//      * next step's u tile prefetched (global_prefetch) during compute
// ---------------------------------------------------------------------------

typedef __attribute__((ext_vector_type(16))) __bf16 v16bf;
typedef __attribute__((ext_vector_type(8)))  __bf16 v8bf;
typedef __attribute__((ext_vector_type(4)))  __bf16 v4bf;
typedef __attribute__((ext_vector_type(8)))  float  v8f;
typedef __attribute__((ext_vector_type(4)))  float  v4f;

#define BB 512   // batch
#define TT 512   // control sequence length
#define SDm 256  // state dim
#define CDm 64   // control dim
#define HHm 512  // rnn hidden
#define D1m 1024
#define D2m 1024

// LDS layout for persistent RNN kernel (padded strides -> conflict-free banks)
#define SW_LD 584   // 576 + 8 bf16 pad (row advance 36 banks; 16 distinct)
#define SA_LD 520   // 512 + 8 bf16 pad (row advance  4 banks; 16 distinct)
#define SU_LD 72    //  64 + 8 bf16 pad (row advance 36 banks; 16 distinct)
#define SMEM_BYTES ((128 * SW_LD + 128 * SA_LD + 128 * SU_LD) * 2)

// --- fp32 -> bf16 conversion (with column slice) ----------------------------
__global__ void cvt_bf16_kernel(const float* __restrict__ src, __bf16* __restrict__ dst,
                                int rows, int cols, int src_ld, int src_col0) {
    int idx = blockIdx.x * 256 + threadIdx.x;
    if (idx >= rows * cols) return;
    int r = idx / cols, c = idx - r * cols;
    dst[idx] = (__bf16)src[(size_t)r * src_ld + src_col0 + c];
}

__global__ void zero_bf16_kernel(__bf16* __restrict__ p, int n) {
    int idx = blockIdx.x * 256 + threadIdx.x;
    if (idx < n) p[idx] = (__bf16)0.0f;
}

__global__ void zero_u32_kernel(unsigned* __restrict__ p, int n) {
    int idx = blockIdx.x * 32 + threadIdx.x;
    if (idx < n) p[idx] = 0u;
}

// --- WMMA fragment loaders --------------------------------------------------
// A operand (16x32 bf16, MxK): lane = half*16 + r holds row m0+r, K chunks
// [k0,k0+8) and [k0+16,k0+24), k0 = kc + half*8.
__device__ __forceinline__ v16bf load_a(const __bf16* A, int lda, int row, int k0) {
    const __bf16* p = A + (size_t)row * lda + k0;
    v8bf b0 = *(const v8bf*)p;
    v8bf b1 = *(const v8bf*)(p + 16);
    return __builtin_shufflevector(b0, b1, 0,1,2,3,4,5,6,7, 8,9,10,11,12,13,14,15);
}
__device__ __forceinline__ v16bf load_a(const float* A, int lda, int row, int k0) {
    const float* p = A + (size_t)row * lda + k0;
    v4f x0 = *(const v4f*)(p);
    v4f x1 = *(const v4f*)(p + 4);
    v4f x2 = *(const v4f*)(p + 16);
    v4f x3 = *(const v4f*)(p + 20);
    v16bf t;
    t[0]=(__bf16)x0[0]; t[1]=(__bf16)x0[1]; t[2]=(__bf16)x0[2]; t[3]=(__bf16)x0[3];
    t[4]=(__bf16)x1[0]; t[5]=(__bf16)x1[1]; t[6]=(__bf16)x1[2]; t[7]=(__bf16)x1[3];
    t[8]=(__bf16)x2[0]; t[9]=(__bf16)x2[1]; t[10]=(__bf16)x2[2]; t[11]=(__bf16)x2[3];
    t[12]=(__bf16)x3[0]; t[13]=(__bf16)x3[1]; t[14]=(__bf16)x3[2]; t[15]=(__bf16)x3[3];
    return t;
}
// B operand (32x16 bf16, KxN): lane = half*16 + r holds column n0+r,
// K chunks [k0,k0+8) and [k0+16,k0+24), k0 = kc + half*8. Row-major W[NxK].
__device__ __forceinline__ v16bf load_b(const __bf16* W, int ldw, int ncol, int k0) {
    const __bf16* q = W + (size_t)ncol * ldw + k0;
    v8bf b0 = *(const v8bf*)q;
    v8bf b1 = *(const v8bf*)(q + 16);
    return __builtin_shufflevector(b0, b1, 0,1,2,3,4,5,6,7, 8,9,10,11,12,13,14,15);
}

// --- K loop over one A segment ----------------------------------------------
template <typename AT>
__device__ __forceinline__ void kloop(const AT* __restrict__ A, int lda, int K,
                                      const __bf16* __restrict__ W, int ldw, int kofs,
                                      int mrow0, int ncol0, int half, int r,
                                      v8f (&acc)[4][2]) {
    for (int kc = 0; kc < K; kc += 32) {
        v16bf af[4];
#pragma unroll
        for (int mi = 0; mi < 4; ++mi)
            af[mi] = load_a(A, lda, mrow0 + mi * 16 + r, kc + half * 8);
        v16bf bfm[2];
#pragma unroll
        for (int ni = 0; ni < 2; ++ni)
            bfm[ni] = load_b(W, ldw, ncol0 + ni * 16 + r, kofs + kc + half * 8);
#pragma unroll
        for (int mi = 0; mi < 4; ++mi)
#pragma unroll
            for (int ni = 0; ni < 2; ++ni)
                acc[mi][ni] = __builtin_amdgcn_wmma_f32_16x16x32_bf16(
                    false, af[mi], false, bfm[ni], (short)0, acc[mi][ni], false, false);
    }
}

// --- One-shot GEMM (MLPs / h2o / combinator) --------------------------------
template <bool TANH, bool RANK1, bool OUT_BF16, typename AT>
__global__ __launch_bounds__(256, 1)
void gemm_wmma_kernel(const AT* __restrict__ A1, int lda1, int K1,
                      const __bf16* __restrict__ A2, int lda2, int K2,
                      const __bf16* __restrict__ W, int ldw,
                      const float* __restrict__ bias,
                      const float* __restrict__ tvec,
                      const float* __restrict__ w0, int w0s,
                      void* __restrict__ outv, int ldo, int ocol0) {
    const int lane = threadIdx.x & 31;
    const int wave = threadIdx.x >> 5;
    const int wm = wave & 1, wn = wave >> 1;
    const int mrow0 = blockIdx.x * 128 + wm * 64;
    const int ncol0 = blockIdx.y * 128 + wn * 32;
    const int half = lane >> 4, r = lane & 15;

    v8f acc[4][2];
#pragma unroll
    for (int mi = 0; mi < 4; ++mi)
#pragma unroll
        for (int ni = 0; ni < 2; ++ni) acc[mi][ni] = (v8f)(0.0f);

    kloop(A1, lda1, K1, W, ldw, 0, mrow0, ncol0, half, r, acc);
    if (K2 > 0) kloop(A2, lda2, K2, W, ldw, K1, mrow0, ncol0, half, r, acc);

#pragma unroll
    for (int ni = 0; ni < 2; ++ni) {
        const int col = ncol0 + ni * 16 + r;
        const float bb = bias[col];
        float w0c = 0.0f;
        if (RANK1) w0c = w0[(size_t)col * w0s];
#pragma unroll
        for (int mi = 0; mi < 4; ++mi) {
#pragma unroll
            for (int i = 0; i < 8; ++i) {
                const int row = mrow0 + mi * 16 + half * 8 + i;
                float v = acc[mi][ni][i] + bb;
                if (RANK1) v += tvec[row] * w0c;
                if (TANH) v = tanhf(v);
                const size_t o = (size_t)row * ldo + ocol0 + col;
                if constexpr (OUT_BF16)
                    ((__bf16*)outv)[o] = (__bf16)v;
                else
                    ((float*)outv)[o] = v;
            }
        }
    }
}

// --- Persistent RNN scan: all 511 hidden updates in one launch --------------
// 16 blocks; block (mb,nb) owns H tile [mb*128,+128)x[nb*128,+128).
// Dependencies are confined to row group mb (4 blocks), so each group has its
// own atomic barrier counter.
__global__ __launch_bounds__(256, 1)
void rnn_persistent_kernel(const float* __restrict__ u,
                           const __bf16* __restrict__ Wb,   // i2h bf16 [512 x 576]
                           const float* __restrict__ bias,  // [512]
                           __bf16* __restrict__ hA, __bf16* __restrict__ hB,
                           unsigned* __restrict__ gcount, int nsteps) {
    extern __shared__ __bf16 smem[];
    __bf16* sW = smem;                                 // [128][SW_LD]
    __bf16* sA = smem + 128 * SW_LD;                   // [128][SA_LD]
    __bf16* sU = smem + 128 * (SW_LD + SA_LD);         // [128][SU_LD]

    const int tid  = threadIdx.x;
    const int lane = tid & 31;
    const int wave = tid >> 5;
    const int wm = wave & 1, wn = wave >> 1;
    const int half = lane >> 4, r = lane & 15;
    const int mb = blockIdx.x >> 2, nb = blockIdx.x & 3;
    const int mloc0 = wm * 64, nloc0 = wn * 32;
    unsigned* gc = gcount + mb;   // row-group barrier counter

    // Stage this block's 128 x 576 weight slice into LDS once (16B chunks).
    for (int i = tid; i < 128 * 72; i += 256) {
        int rr = i / 72, cc = (i - rr * 72) * 8;
        *(v8bf*)(sW + rr * SW_LD + cc) =
            *(const v8bf*)(Wb + (size_t)(nb * 128 + rr) * (CDm + HHm) + cc);
    }

#pragma unroll 1
    for (int step = 0; step < nsteps; ++step) {
        const __bf16* Hsrc = (step & 1) ? hB : hA;
        __bf16* Hdst = (step & 1) ? hA : hB;
        const float* ut = u + (size_t)step * BB * CDm + (size_t)mb * 128 * CDm;

        // Stage A rows [mb*128, +128) x 512 of H_prev into LDS.
        for (int i = tid; i < 128 * 64; i += 256) {
            int rr = i >> 6, cc = (i & 63) * 8;
            *(v8bf*)(sA + rr * SA_LD + cc) =
                *(const v8bf*)(Hsrc + (size_t)(mb * 128 + rr) * HHm + cc);
        }
        // Stage u_t tile (128 x 64 fp32) into LDS as bf16: one cvt per element.
        for (int i = tid; i < 128 * 16; i += 256) {
            int rr = i >> 4, cc = (i & 15) * 4;
            v4f f = *(const v4f*)(ut + (size_t)rr * CDm + cc);
            v4bf o;
            o[0] = (__bf16)f[0]; o[1] = (__bf16)f[1];
            o[2] = (__bf16)f[2]; o[3] = (__bf16)f[3];
            *(v4bf*)(sU + rr * SU_LD + cc) = o;
        }
        // Prefetch next step's u tile (always a valid address: step+1 <= T-1).
        {
            const float* utn = u + (size_t)(step + 1) * BB * CDm + (size_t)mb * 128 * CDm;
            for (int i = tid; i < 512; i += 256)            // 512 x 64B lines = 32KB
                __builtin_prefetch(utn + (size_t)i * 16, 0, 1);
        }
        __syncthreads();

        v8f acc[4][2];
#pragma unroll
        for (int mi = 0; mi < 4; ++mi)
#pragma unroll
            for (int ni = 0; ni < 2; ++ni) acc[mi][ni] = (v8f)(0.0f);

        kloop((const __bf16*)sU, SU_LD, CDm, sW, SW_LD, 0, mloc0, nloc0, half, r, acc);
        kloop((const __bf16*)sA, SA_LD, HHm, sW, SW_LD, CDm, mloc0, nloc0, half, r, acc);

#pragma unroll
        for (int ni = 0; ni < 2; ++ni) {
            const int col = nb * 128 + nloc0 + ni * 16 + r;
            const float bb = bias[col];
#pragma unroll
            for (int mi = 0; mi < 4; ++mi) {
#pragma unroll
                for (int i = 0; i < 8; ++i) {
                    const int row = mb * 128 + mloc0 + mi * 16 + half * 8 + i;
                    float v = tanhf(acc[mi][ni][i] + bb);
                    Hdst[(size_t)row * HHm + col] = (__bf16)v;
                }
            }
        }

        // Row-group barrier (4 blocks): release tile, arrive, spin, acquire.
        __threadfence();
        __syncthreads();
        if (tid == 0) {
            __hip_atomic_fetch_add(gc, 1u, __ATOMIC_RELEASE, __HIP_MEMORY_SCOPE_AGENT);
            const unsigned target = 4u * (unsigned)(step + 1);
            while (__hip_atomic_load(gc, __ATOMIC_ACQUIRE, __HIP_MEMORY_SCOPE_AGENT) < target)
                __builtin_amdgcn_s_sleep(1);
        }
        __syncthreads();
        __threadfence();
    }
}

// ---------------------------------------------------------------------------
extern "C" void kernel_launch(void* const* d_in, const int* in_sizes, int n_in,
                              void* d_out, int out_size, void* d_ws, size_t ws_size,
                              hipStream_t stream) {
    (void)in_sizes; (void)n_in; (void)out_size; (void)ws_size;

    const float* t     = (const float*)d_in[0];
    const float* x     = (const float*)d_in[1];
    const float* u     = (const float*)d_in[2];
    const float* i2h_W = (const float*)d_in[3];
    const float* i2h_b = (const float*)d_in[4];
    const float* h2o_W = (const float*)d_in[5];
    const float* h2o_b = (const float*)d_in[6];
    const float* x1_W  = (const float*)d_in[7];
    const float* x1_b  = (const float*)d_in[8];
    const float* x2_W  = (const float*)d_in[9];
    const float* x2_b  = (const float*)d_in[10];
    const float* x3_W  = (const float*)d_in[11];
    const float* x3_b  = (const float*)d_in[12];
    const float* u1_W  = (const float*)d_in[13];
    const float* u1_b  = (const float*)d_in[14];
    const float* u2_W  = (const float*)d_in[15];
    const float* u2_b  = (const float*)d_in[16];
    const float* u3_W  = (const float*)d_in[17];
    const float* u3_b  = (const float*)d_in[18];
    const float* comb_W = (const float*)d_in[19];
    const float* comb_b = (const float*)d_in[20];

    char* ws = (char*)d_ws;
    size_t off = 0;
    auto alloc = [&](size_t elems) -> __bf16* {
        __bf16* p = (__bf16*)(ws + off);
        off += ((elems * sizeof(__bf16)) + 255) & ~(size_t)255;
        return p;
    };

    __bf16* i2hb  = alloc((size_t)HHm * (CDm + HHm));
    __bf16* h2ob  = alloc((size_t)SDm * (CDm + HHm));
    __bf16* x1b   = alloc((size_t)D1m * SDm);
    __bf16* x2b   = alloc((size_t)D2m * D1m);
    __bf16* x3b   = alloc((size_t)SDm * D2m);
    __bf16* u1b   = alloc((size_t)D1m * SDm);
    __bf16* u2b   = alloc((size_t)D2m * D1m);
    __bf16* u3b   = alloc((size_t)SDm * D2m);
    __bf16* combb = alloc((size_t)SDm * (2 * SDm));
    __bf16* xb    = alloc((size_t)BB * SDm);
    __bf16* hA    = alloc((size_t)BB * HHm);
    __bf16* hB    = alloc((size_t)BB * HHm);
    __bf16* urnnb = alloc((size_t)BB * SDm);
    __bf16* h1b   = alloc((size_t)BB * D1m);
    __bf16* h2b   = alloc((size_t)BB * D2m);
    __bf16* cinb  = alloc((size_t)BB * (2 * SDm));
    unsigned* gcount = (unsigned*)alloc(128);

    const dim3 blk(256);
    auto cvt = [&](const float* s, __bf16* d, int rows, int cols, int ld, int c0) {
        int n = rows * cols;
        cvt_bf16_kernel<<<dim3((n + 255) / 256), blk, 0, stream>>>(s, d, rows, cols, ld, c0);
    };

    // One-time fp32 -> bf16 conversions (t-column sliced off layer-1 weights).
    cvt(i2h_W, i2hb, HHm, CDm + HHm, CDm + HHm, 0);
    cvt(h2o_W, h2ob, SDm, CDm + HHm, CDm + HHm, 0);
    cvt(x1_W,  x1b,  D1m, SDm, 1 + SDm, 1);
    cvt(x2_W,  x2b,  D2m, D1m, D1m, 0);
    cvt(x3_W,  x3b,  SDm, D2m, D2m, 0);
    cvt(u1_W,  u1b,  D1m, SDm, 1 + SDm, 1);
    cvt(u2_W,  u2b,  D2m, D1m, D1m, 0);
    cvt(u3_W,  u3b,  SDm, D2m, D2m, 0);
    cvt(comb_W, combb, SDm, 2 * SDm, 2 * SDm, 0);
    cvt(x, xb, BB, SDm, SDm, 0);
    {
        int n = BB * HHm;
        zero_bf16_kernel<<<dim3((n + 255) / 256), blk, 0, stream>>>(hA, n);
    }
    zero_u32_kernel<<<dim3(1), dim3(32), 0, stream>>>(gcount, 4);

    // ---- x_dnn: tanh(L1) -> tanh(L2) -> L3 -> cinb[:, 0:256]
    gemm_wmma_kernel<true, true, true, __bf16><<<dim3(BB / 128, D1m / 128), blk, 0, stream>>>(
        xb, SDm, SDm, nullptr, 0, 0, x1b, SDm, x1_b, t, x1_W, 1 + SDm, h1b, D1m, 0);
    gemm_wmma_kernel<true, false, true, __bf16><<<dim3(BB / 128, D2m / 128), blk, 0, stream>>>(
        h1b, D1m, D1m, nullptr, 0, 0, x2b, D1m, x2_b, nullptr, nullptr, 0, h2b, D2m, 0);
    gemm_wmma_kernel<false, false, true, __bf16><<<dim3(BB / 128, SDm / 128), blk, 0, stream>>>(
        h2b, D2m, D2m, nullptr, 0, 0, x3b, D2m, x3_b, nullptr, nullptr, 0, cinb, 2 * SDm, 0);

    // ---- recurrent scan: 511 hidden updates in ONE persistent launch.
    // nsteps = 511 (odd): final hidden state lands in hB.
    rnn_persistent_kernel<<<dim3(16), blk, SMEM_BYTES, stream>>>(
        u, i2hb, i2h_b, hA, hB, gcount, TT - 1);

    // ---- final step through h2o on u[T-1]
    gemm_wmma_kernel<true, false, true, float><<<dim3(BB / 128, SDm / 128), blk, 0, stream>>>(
        u + (size_t)(TT - 1) * BB * CDm, CDm, CDm, hB, HHm, HHm, h2ob, CDm + HHm, h2o_b,
        nullptr, nullptr, 0, urnnb, SDm, 0);

    // ---- u_dnn: tanh(L1) -> tanh(L2) -> L3 -> cinb[:, 256:512]
    gemm_wmma_kernel<true, true, true, __bf16><<<dim3(BB / 128, D1m / 128), blk, 0, stream>>>(
        urnnb, SDm, SDm, nullptr, 0, 0, u1b, SDm, u1_b, t, u1_W, 1 + SDm, h1b, D1m, 0);
    gemm_wmma_kernel<true, false, true, __bf16><<<dim3(BB / 128, D2m / 128), blk, 0, stream>>>(
        h1b, D1m, D1m, nullptr, 0, 0, u2b, D1m, u2_b, nullptr, nullptr, 0, h2b, D2m, 0);
    gemm_wmma_kernel<false, false, true, __bf16><<<dim3(BB / 128, SDm / 128), blk, 0, stream>>>(
        h2b, D2m, D2m, nullptr, 0, 0, u3b, D2m, u3_b, nullptr, nullptr, 0, cinb, 2 * SDm, SDm);

    // ---- combinator -> d_out [512 x 256] fp32
    gemm_wmma_kernel<false, false, false, __bf16><<<dim3(BB / 128, SDm / 128), blk, 0, stream>>>(
        cinb, 2 * SDm, 2 * SDm, nullptr, 0, 0, combb, 2 * SDm, comb_b,
        nullptr, nullptr, 0, d_out, SDm, 0);
}
